// FilterbankAttention_24189255811116
// MI455X (gfx1250) — matrix operand order repaired and checked by reading, not compiled, not run on hardware
//
#include <hip/hip_runtime.h>

typedef __attribute__((ext_vector_type(16))) __bf16        v16bf;
typedef __attribute__((ext_vector_type(8)))  float         v8f;
typedef __attribute__((ext_vector_type(4)))  float         v4f;
typedef __attribute__((ext_vector_type(4)))  unsigned int  u4v;

#define IMG_A 256
#define IMG_B 256
#define FB_N  64

// ---------------------------------------------------------------------------
// Prep kernel: params -> F_X (bf16) and gamma*F_Y (bf16) in workspace.
// One block of 256 threads; cost is negligible vs. the 256MB main pass.
// ---------------------------------------------------------------------------
__global__ __launch_bounds__(256) void fb_prepare(
    const float* __restrict__ h, const float* __restrict__ Ww,
    const float* __restrict__ Wb, __bf16* __restrict__ FX,
    __bf16* __restrict__ FYg)
{
  __shared__ float red[256];
  __shared__ float par[5];
  __shared__ float sums[2];
  const int t = threadIdx.x;

  // params[j] = dot(h[0,:], W[j,:]) + b[j]
  for (int j = 0; j < 5; ++j) {
    float s = 0.f;
    for (int i = t; i < 1024; i += 256) s += h[i] * Ww[j * 1024 + i];
    red[t] = s; __syncthreads();
    for (int off = 128; off > 0; off >>= 1) {
      if (t < off) red[t] += red[t + off];
      __syncthreads();
    }
    if (t == 0) par[j] = red[0] + Wb[j];
    __syncthreads();
  }

  const float var   = expf(par[2] + 1e-8f);
  const float dt    = expf(par[3]);
  const float gamma = expf(par[4]);
  const float gX    = 0.5f * (IMG_A + 1) * (par[0] + 1.f);
  const float gY    = 0.5f * (IMG_B + 1) * (par[1] + 1.f);
  const float d     = dt * (float)(IMG_A - 1) / (float)(FB_N - 1);
  const float sub   = (-FB_N * 0.5f - 0.5f) * d;   // -32.5 * d
  const float inv2v = 1.f / (2.f * var);

  // pass 1: global sums of the unnormalized filterbanks
  float sx = 0.f, sy = 0.f;
  for (int i = t; i < FB_N * IMG_A; i += 256) {
    const int   n   = i >> 8;
    const float col = (float)(i & 255);
    const float mu  = sub + d * (float)n;
    const float dx  = col - (gX + mu);
    const float dy  = col - (gY + mu);
    sx += expf(-dx * dx * inv2v);
    sy += expf(-dy * dy * inv2v);
  }
  red[t] = sx; __syncthreads();
  for (int off = 128; off > 0; off >>= 1) { if (t < off) red[t] += red[t + off]; __syncthreads(); }
  if (t == 0) sums[0] = red[0];
  __syncthreads();
  red[t] = sy; __syncthreads();
  for (int off = 128; off > 0; off >>= 1) { if (t < off) red[t] += red[t + off]; __syncthreads(); }
  if (t == 0) sums[1] = red[0];
  __syncthreads();

  const float invSX  = 1.f / sums[0];
  const float invSYg = gamma / sums[1];   // fold gamma into F_Y

  // pass 2: write normalized bf16 filterbanks
  for (int i = t; i < FB_N * IMG_A; i += 256) {
    const int   n   = i >> 8;
    const float col = (float)(i & 255);
    const float mu  = sub + d * (float)n;
    const float dx  = col - (gX + mu);
    const float dy  = col - (gY + mu);
    FX[i]  = (__bf16)(expf(-dx * dx * inv2v) * invSX);
    FYg[i] = (__bf16)(expf(-dy * dy * inv2v) * invSYg);
  }
}

// ---------------------------------------------------------------------------
// Main kernel: per image  out = (gamma*F_Y) @ im @ F_X^T  via bf16 WMMA.
// Grid: 1024 blocks (512 x-images, then 512 x_hat-images), 256 threads.
// ---------------------------------------------------------------------------
#define GT_PITCH 264   // 256 + 8 bf16 pad: staggers LDS banks per row

__global__ __launch_bounds__(256) void fb_main(
    const float* __restrict__ x, const float* __restrict__ xh,
    const __bf16* __restrict__ FX, const __bf16* __restrict__ FYg,
    float* __restrict__ out)
{
  __shared__ __bf16 GT[FB_N * GT_PITCH];   // G^T[m][a], bf16, 33 KB

  const int tid  = threadIdx.x;
  const int wave = tid >> 5;       // 0..7
  const int lane = tid & 31;
  const int half = (lane >> 4) & 1;
  const int l16  = lane & 15;
  const int img  = blockIdx.x;

  const float* src = (img < 512) ? (x  + (size_t)img * (IMG_A * IMG_B))
                                 : (xh + (size_t)(img - 512) * (IMG_A * IMG_B));
  float* dst = out + (size_t)(img & 511) * (2 * FB_N * FB_N)
                   + ((img < 512) ? 0 : FB_N * FB_N);

  // ---------------- Stage 1: G[a,m] = sum_b im[a,b] * FX[m,b] -------------
  // wave owns image-row tiles {2w, 2w+1}, all 4 m-tiles.
  v8f acc[2][4];
#pragma unroll
  for (int j = 0; j < 2; ++j)
#pragma unroll
    for (int nt = 0; nt < 4; ++nt) acc[j][nt] = (v8f){0,0,0,0,0,0,0,0};

  for (int k = 0; k < 8; ++k) {                 // K = 256 in steps of 32
    const int kc = k * 32;
    // A operands: row a = Mtile*16 + l16; K layout: {8h..8h+7} U {16+8h..23+8h}
    v16bf Aop[2];
#pragma unroll
    for (int j = 0; j < 2; ++j) {
      const float* rp = src + (size_t)((2 * wave + j) * 16 + l16) * IMG_B
                            + kc + 8 * half;
      const v4f r0 = *reinterpret_cast<const v4f*>(rp);
      const v4f r1 = *reinterpret_cast<const v4f*>(rp + 4);
      const v4f r2 = *reinterpret_cast<const v4f*>(rp + 16);
      const v4f r3 = *reinterpret_cast<const v4f*>(rp + 20);
#pragma unroll
      for (int e = 0; e < 4; ++e) {
        Aop[j][e]      = (__bf16)r0[e];
        Aop[j][e + 4]  = (__bf16)r1[e];
        Aop[j][e + 8]  = (__bf16)r2[e];
        Aop[j][e + 12] = (__bf16)r3[e];
      }
    }
    // B operands: col m = nt*16 + l16; K layout: {8h..8h+7} U {16+8h..23+8h}
#pragma unroll
    for (int nt = 0; nt < 4; ++nt) {
      union { u4v u[2]; v16bf v; } bv;
      const u4v* bp = reinterpret_cast<const u4v*>(
          FX + (size_t)(nt * 16 + l16) * IMG_B + kc + 8 * half);
      bv.u[0] = bp[0];
      bv.u[1] = bp[2];
#pragma unroll
      for (int j = 0; j < 2; ++j)
        acc[j][nt] = __builtin_amdgcn_wmma_f32_16x16x32_bf16(
            false, Aop[j], false, bv.v, (short)0, acc[j][nt], false, false);
    }
  }

  // write G^T to LDS as bf16: lane's 8 accum elems = 8 consecutive columns
#pragma unroll
  for (int j = 0; j < 2; ++j)
#pragma unroll
    for (int nt = 0; nt < 4; ++nt) {
      union { __bf16 b[8]; u4v u; } pk;
#pragma unroll
      for (int v = 0; v < 8; ++v) pk.b[v] = (__bf16)acc[j][nt][v];
      *reinterpret_cast<u4v*>(
          &GT[(size_t)(nt * 16 + l16) * GT_PITCH + (2 * wave + j) * 16 + 8 * half]) = pk.u;
    }
  __syncthreads();

  // ---------------- Stage 2: out[n,m] = sum_a FYg[n,a] * G[a,m] -----------
  const int ntile = wave & 3;
  const int mt0   = (wave >> 2) * 2;       // m-tiles {mt0, mt0+1}
  v8f oacc[2];
  oacc[0] = (v8f){0,0,0,0,0,0,0,0};
  oacc[1] = (v8f){0,0,0,0,0,0,0,0};

  for (int k = 0; k < 8; ++k) {
    const int kc = k * 32;
    union { u4v u[2]; v16bf v; } av;
    const __bf16* ap = FYg + (size_t)(ntile * 16 + l16) * IMG_A + kc + 8 * half;
    av.u[0] = *reinterpret_cast<const u4v*>(ap);
    av.u[1] = *reinterpret_cast<const u4v*>(ap + 16);
#pragma unroll
    for (int j = 0; j < 2; ++j) {
      union { u4v u[2]; v16bf v; } bv;
      const __bf16* bp = &GT[(size_t)((mt0 + j) * 16 + l16) * GT_PITCH + kc + 8 * half];
      bv.u[0] = *reinterpret_cast<const u4v*>(bp);
      bv.u[1] = *reinterpret_cast<const u4v*>(bp + 16);
      oacc[j] = __builtin_amdgcn_wmma_f32_16x16x32_bf16(
          false, av.v, false, bv.v, (short)0, oacc[j], false, false);
    }
  }

  // D layout: row n = ntile*16 + v + 8*half, col m = mtile*16 + l16
#pragma unroll
  for (int j = 0; j < 2; ++j) {
    const int m = (mt0 + j) * 16 + l16;
#pragma unroll
    for (int v = 0; v < 8; ++v) {
      const int n = ntile * 16 + 8 * half + v;
      dst[n * FB_N + m] = oacc[j][v];
    }
  }
}

// ---------------------------------------------------------------------------
extern "C" void kernel_launch(void* const* d_in, const int* in_sizes, int n_in,
                              void* d_out, int out_size, void* d_ws, size_t ws_size,
                              hipStream_t stream) {
  (void)in_sizes; (void)n_in; (void)out_size; (void)ws_size;
  const float* x   = (const float*)d_in[0];   // [512, 65536]
  const float* xh  = (const float*)d_in[1];   // [512, 65536]
  const float* h   = (const float*)d_in[2];   // [512, 1024]
  const float* Ww  = (const float*)d_in[3];   // [5, 1024]
  const float* Wb  = (const float*)d_in[4];   // [5]
  float* out = (float*)d_out;                 // [512, 8192]

  __bf16* FX  = reinterpret_cast<__bf16*>(d_ws);      // [64,256] bf16
  __bf16* FYg = FX + FB_N * IMG_A;                    // [64,256] bf16 (gamma folded)

  fb_prepare<<<1, 256, 0, stream>>>(h, Ww, Wb, FX, FYg);
  fb_main<<<1024, 256, 0, stream>>>(x, xh, FX, FYg, out);
}
